// CrossAttention_53541062312468
// MI455X (gfx1250) — matrix operand order repaired and checked by reading, not compiled, not run on hardware
//
#include <hip/hip_runtime.h>

// ---------------------------------------------------------------------------
// CrossAttention for MI455X (gfx1250). Compute-bound (~69 GFLOP, working set
// L2-resident), everything through v_wmma_f32_16x16x32_f16.
//   Stage 0: cvt_kernel  : m1..m4, Wqkv, Wproj -> f16 workspace (once)
//   Stage 1: qkv_kernel  : Wqkv16 @ X16^T (transposed tiles) ->
//                          Q (f16 [i][b][h][n][d], pre-scaled by D^-0.5),
//                          K (f16 [i][b][h][n][d]), Vt (f16 [i][b][h][d][n])
//   Stage 2: attn_kernel : M=32 blocked flash attention (no-max softmax,
//                          deferred row-sum; K/Vt fragments reused across two
//                          q-tiles), sum over j!=i -> Xattn (f16)
//   Stage 3: proj_kernel : Wproj16 @ Xattn^T + 3*bproj -> out (f32 [4][B][N][C])
// ---------------------------------------------------------------------------

typedef __attribute__((ext_vector_type(16))) _Float16 v16h;
typedef __attribute__((ext_vector_type(8)))  _Float16 v8h;
typedef __attribute__((ext_vector_type(4)))  _Float16 v4h;
typedef __attribute__((ext_vector_type(8)))  float    v8f;
typedef __attribute__((ext_vector_type(4)))  float    v4f;

#define WMMA_F16(a, b, c) \
  __builtin_amdgcn_wmma_f32_16x16x32_f16(false, (a), false, (b), (short)0, (c), false, false)

#define BB 2
#define NN 1024
#define CC 512
#define HH 8
#define DD 64
#define SCALE 0.125f  // 64^-0.5, folded into Q at stage-1 store

// ---------------------------------------------------------------------------
// Stage 0: one-time f32 -> f16 conversion (vectorized 4-wide).
// ---------------------------------------------------------------------------
__global__ __launch_bounds__(256) void cvt_kernel(
    const float* __restrict__ s, _Float16* __restrict__ d, int n4)
{
  int t = blockIdx.x * 256 + threadIdx.x;
  if (t >= n4) return;
  v4f v = ((const v4f*)s)[t];
  v4h o;
  #pragma unroll
  for (int e = 0; e < 4; ++e) o[e] = (_Float16)v[e];
  ((v4h*)d)[t] = o;
}

// ---------------------------------------------------------------------------
// Stage 1: QKV projection, transposed tiles (A = Wqkv rows, B = X rows).
// One wave -> 32(o) x 32(n) tile. Grid: 4*2*48*32 = 12288 waves.
// ---------------------------------------------------------------------------
__global__ __launch_bounds__(256) void qkv_kernel(
    const _Float16* __restrict__ X16, const _Float16* __restrict__ W16,
    _Float16* __restrict__ Q, _Float16* __restrict__ K, _Float16* __restrict__ Vt)
{
  int wave = (blockIdx.x * blockDim.x + threadIdx.x) >> 5;
  wave = __builtin_amdgcn_readfirstlane(wave);        // wave-uniform -> SGPR
  int lane = threadIdx.x & 31;

  int nt = wave & 31;  wave >>= 5;
  int ot = wave % 48;  wave /= 48;
  int b  = wave & 1;   wave >>= 1;
  int i  = wave;

  int n0 = nt * 32;
  int o0 = ot * 32;

  int lm  = lane & 15;
  int abk = (lane >> 4) << 3;
  int bbk = (lane >> 4) << 4;
  int rowoff = (lane >> 4) << 3;

  const _Float16* Wr = W16 + (size_t)i * 3 * CC * CC + (size_t)(o0 + lm) * CC;
  const _Float16* Xr = X16 + (((size_t)i * BB + b) * NN + n0 + lm) * CC;

  v8f acc[2][2];
  #pragma unroll
  for (int of = 0; of < 2; ++of)
    #pragma unroll
    for (int nf = 0; nf < 2; ++nf)
      #pragma unroll
      for (int r = 0; r < 8; ++r) acc[of][nf][r] = 0.0f;

  for (int k0 = 0; k0 < CC; k0 += 32) {
    __builtin_prefetch(Xr + k0 + 32, 0, 0);
    v16h a[2], bf[2];
    #pragma unroll
    for (int of = 0; of < 2; ++of)
      #pragma unroll
      for (int hh = 0; hh < 16; ++hh) {
        int ka = k0 + bbk + hh;
        a[of][hh] = Wr[(size_t)of * 16 * CC + ka];
      }
    #pragma unroll
    for (int nf = 0; nf < 2; ++nf)
      #pragma unroll
      for (int hh = 0; hh < 16; ++hh)
        bf[nf][hh] = Xr[(size_t)nf * 16 * CC + k0 + bbk + hh];
    #pragma unroll
    for (int of = 0; of < 2; ++of)
      #pragma unroll
      for (int nf = 0; nf < 2; ++nf)
        acc[of][nf] = WMMA_F16(a[of], bf[nf], acc[of][nf]);
  }

  #pragma unroll
  for (int of = 0; of < 2; ++of) {
    int oc = o0 + of * 16;            // SGPR
    int s  = oc >> 9;                 // 0:q 1:k 2:v
    int hd = (oc & 511) >> 6;
    int d0 = oc & 63;
    size_t base = (((size_t)i * BB + b) * HH + hd) * (size_t)NN * DD;
    if (s < 2) {
      _Float16* dst = (s == 0) ? Q : K;
      float mul = (s == 0) ? SCALE : 1.0f;
      #pragma unroll
      for (int nf = 0; nf < 2; ++nf) {
        int n = n0 + nf * 16 + lm;
        v8h v;
        #pragma unroll
        for (int r = 0; r < 8; ++r) v[r] = (_Float16)(acc[of][nf][r] * mul);
        *(v8h*)(dst + base + (size_t)n * DD + d0 + rowoff) = v;   // 16B store
      }
    } else {
      #pragma unroll
      for (int nf = 0; nf < 2; ++nf) {
        int n = n0 + nf * 16 + lm;
        #pragma unroll
        for (int r = 0; r < 8; ++r)
          Vt[base + (size_t)(d0 + r + rowoff) * NN + n] = (_Float16)acc[of][nf][r];
      }
    }
  }
}

// ---------------------------------------------------------------------------
// Stage 2: fused cross attention, M=32 per wave (2 q-tiles). K/Vt B-fragments
// are loaded once per 32-key step and reused for both q-tiles: 16 WMMAs per
// 16 b128 B-loads. No-max softmax (scale folded into Q), row-sum deferred to
// one reduction per j. Grid: 4*2*8*32 = 2048 waves -> 256 blocks.
// ---------------------------------------------------------------------------
__global__ __launch_bounds__(256) void attn_kernel(
    const _Float16* __restrict__ Q, const _Float16* __restrict__ K,
    const _Float16* __restrict__ Vt, _Float16* __restrict__ Xout)
{
  __shared__ _Float16 ldsP[8][2][16 * 32];   // per-wave, per-q-tile (16 KB)

  int wid  = threadIdx.x >> 5;
  int lane = threadIdx.x & 31;
  int gw   = __builtin_amdgcn_readfirstlane(blockIdx.x * 8 + wid);

  int mt = gw & 31; gw >>= 5;     // 32 m-tiles of 32 rows
  int h  = gw & 7;  gw >>= 3;
  int b  = gw & 1;  gw >>= 1;
  int i  = gw;
  int m0 = mt * 32;

  int lm  = lane & 15;
  int abk = (lane >> 4) << 3;
  int bbk = (lane >> 4) << 4;
  int rowoff = (lane >> 4) << 3;

  // Q fragments (pre-scaled): qa[q-tile][k-frag]
  size_t qbase = ((((size_t)i * BB + b) * HH + h) * NN + m0) * DD;
  v16h qa[2][2];
  #pragma unroll
  for (int qt = 0; qt < 2; ++qt)
    #pragma unroll
    for (int f = 0; f < 2; ++f)
      #pragma unroll
      for (int hh = 0; hh < 16; ++hh) {
        int ka = f * 32 + bbk + hh;
        qa[qt][f][hh] = Q[qbase + (size_t)(qt * 16 + lm) * DD + ka];
      }

  v8f tot[2][4];
  #pragma unroll
  for (int qt = 0; qt < 2; ++qt)
    #pragma unroll
    for (int f = 0; f < 4; ++f)
      #pragma unroll
      for (int r = 0; r < 8; ++r) tot[qt][f][r] = 0.0f;

  for (int jj = 0; jj < 3; ++jj) {
    int j = jj + (jj >= i);   // skip i
    size_t kvbase = (((size_t)j * BB + b) * HH + h) * (size_t)NN * DD;

    v8f acc[2][4];
    float psum[2][8];
    #pragma unroll
    for (int qt = 0; qt < 2; ++qt) {
      #pragma unroll
      for (int f = 0; f < 4; ++f)
        #pragma unroll
        for (int r = 0; r < 8; ++r) acc[qt][f][r] = 0.0f;
      #pragma unroll
      for (int r = 0; r < 8; ++r) psum[qt][r] = 0.0f;
    }

    for (int kk = 0; kk < NN; kk += 32) {
      __builtin_prefetch(K + kvbase + (size_t)(kk + 32 + lm) * DD, 0, 0);

      // ---- S = Q*K^T: B-fragments shared across both q-tiles (8 WMMAs) ----
      v8f s0[2], s1[2];
      #pragma unroll
      for (int qt = 0; qt < 2; ++qt)
        #pragma unroll
        for (int r = 0; r < 8; ++r) { s0[qt][r] = 0.0f; s1[qt][r] = 0.0f; }
      #pragma unroll
      for (int f = 0; f < 2; ++f) {
        v16h bk0, bk1;
        #pragma unroll
        for (int hh = 0; hh < 16; ++hh) {
          int kd = f * 32 + bbk + hh;                       // 32B run
          bk0[hh] = K[kvbase + (size_t)(kk + lm) * DD + kd];
          bk1[hh] = K[kvbase + (size_t)(kk + 16 + lm) * DD + kd];
        }
        #pragma unroll
        for (int qt = 0; qt < 2; ++qt) {
          s0[qt] = WMMA_F16(qa[qt][f], bk0, s0[qt]);
          s1[qt] = WMMA_F16(qa[qt][f], bk1, s1[qt]);
        }
      }

      // ---- unnormalized softmax + LDS transpose (per q-tile) ----
      v16h pa[2];
      #pragma unroll
      for (int qt = 0; qt < 2; ++qt) {
        _Float16* lp = ldsP[wid][qt];
        #pragma unroll
        for (int r = 0; r < 8; ++r) {
          float p0 = __expf(s0[qt][r]);
          float p1 = __expf(s1[qt][r]);
          psum[qt][r] += p0 + p1;
          int row = r + rowoff;
          lp[row * 32 + lm]      = (_Float16)p0;
          lp[row * 32 + 16 + lm] = (_Float16)p1;
        }
        #pragma unroll
        for (int hh = 0; hh < 16; ++hh) {
          int kp = bbk + hh;
          pa[qt][hh] = lp[lm * 32 + kp];
        }
      }

      // ---- acc += P*V: Vt B-fragments shared across q-tiles (8 WMMAs) ----
      #pragma unroll
      for (int f = 0; f < 4; ++f) {
        v16h bv;
        #pragma unroll
        for (int hh = 0; hh < 16; ++hh) {
          int d = f * 16 + lm;
          bv[hh] = Vt[kvbase + (size_t)d * NN + kk + bbk + hh];  // 32B run
        }
        #pragma unroll
        for (int qt = 0; qt < 2; ++qt)
          acc[qt][f] = WMMA_F16(pa[qt], bv, acc[qt][f]);
      }
    }

    // One cross-lane row-sum reduction per j per q-tile, then accumulate.
    #pragma unroll
    for (int qt = 0; qt < 2; ++qt) {
      float rl[8];
      #pragma unroll
      for (int r = 0; r < 8; ++r) {
        float t = psum[qt][r];
        #pragma unroll
        for (int msk = 1; msk < 16; msk <<= 1)
          t += __shfl_xor(t, msk, 32);
        rl[r] = 1.0f / t;
      }
      #pragma unroll
      for (int f = 0; f < 4; ++f)
        #pragma unroll
        for (int r = 0; r < 8; ++r) tot[qt][f][r] += acc[qt][f][r] * rl[r];
    }
  }

  // write fused attention output Xattn[i][b][n][h*64+d] as f16
  size_t xbase = (((size_t)i * BB + b) * NN) * (size_t)CC;
  int cbase = h * DD;
  #pragma unroll
  for (int qt = 0; qt < 2; ++qt)
    #pragma unroll
    for (int f = 0; f < 4; ++f) {
      int col = cbase + f * 16 + lm;
      #pragma unroll
      for (int r = 0; r < 8; ++r) {
        int row = m0 + qt * 16 + r + rowoff;
        Xout[xbase + (size_t)row * CC + col] = (_Float16)tot[qt][f][r];
      }
    }
}

// ---------------------------------------------------------------------------
// Stage 3: output projection, transposed tiles (A = Wproj rows, B = Xattn).
// Grid: 4*2*32(n-tiles)*16(o-tiles) = 4096 waves.
// ---------------------------------------------------------------------------
__global__ __launch_bounds__(256) void proj_kernel(
    const _Float16* __restrict__ X, const _Float16* __restrict__ W16,
    const float* __restrict__ bproj, float* __restrict__ out)
{
  int wave = (blockIdx.x * blockDim.x + threadIdx.x) >> 5;
  wave = __builtin_amdgcn_readfirstlane(wave);
  int lane = threadIdx.x & 31;

  int ot = wave & 15; wave >>= 4;
  int nt = wave & 31; wave >>= 5;
  int b  = wave & 1;  wave >>= 1;
  int i  = wave;

  int n0 = nt * 32;
  int o0 = ot * 32;

  int lm  = lane & 15;
  int abk = (lane >> 4) << 3;
  int bbk = (lane >> 4) << 4;
  int rowoff = (lane >> 4) << 3;

  const _Float16* Wr = W16 + (size_t)i * CC * CC + (size_t)(o0 + lm) * CC;
  const _Float16* Xr = X   + (((size_t)i * BB + b) * NN + n0 + lm) * CC;

  v8f acc[2][2];
  #pragma unroll
  for (int of = 0; of < 2; ++of)
    #pragma unroll
    for (int nf = 0; nf < 2; ++nf)
      #pragma unroll
      for (int r = 0; r < 8; ++r) acc[of][nf][r] = 0.0f;

  for (int k0 = 0; k0 < CC; k0 += 32) {
    __builtin_prefetch(Xr + k0 + 32, 0, 0);
    v16h a[2], bf[2];
    #pragma unroll
    for (int of = 0; of < 2; ++of)
      #pragma unroll
      for (int hh = 0; hh < 16; ++hh) {
        int ka = k0 + bbk + hh;
        a[of][hh] = Wr[(size_t)of * 16 * CC + ka];
      }
    #pragma unroll
    for (int nf = 0; nf < 2; ++nf)
      #pragma unroll
      for (int hh = 0; hh < 16; ++hh)
        bf[nf][hh] = Xr[(size_t)nf * 16 * CC + k0 + bbk + hh];
    #pragma unroll
    for (int of = 0; of < 2; ++of)
      #pragma unroll
      for (int nf = 0; nf < 2; ++nf)
        acc[of][nf] = WMMA_F16(a[of], bf[nf], acc[of][nf]);
  }

  size_t obase = (((size_t)i * BB + b) * NN) * (size_t)CC;
  #pragma unroll
  for (int of = 0; of < 2; ++of) {
    const float* bp = bproj + (size_t)i * CC + o0 + of * 16 + rowoff;
    v4f b0 = *(const v4f*)bp;
    v4f b1 = *(const v4f*)(bp + 4);
    #pragma unroll
    for (int nf = 0; nf < 2; ++nf) {
      int n = n0 + nf * 16 + lm;
      float* op = out + obase + (size_t)n * CC + o0 + of * 16 + rowoff;
      v4f lo, hi;
      #pragma unroll
      for (int r = 0; r < 4; ++r) {
        lo[r] = acc[of][nf][r]     + 3.0f * b0[r];
        hi[r] = acc[of][nf][4 + r] + 3.0f * b1[r];
      }
      *(v4f*)op       = lo;
      *(v4f*)(op + 4) = hi;
    }
  }
}

// ---------------------------------------------------------------------------
extern "C" void kernel_launch(void* const* d_in, const int* in_sizes, int n_in,
                              void* d_out, int out_size, void* d_ws, size_t ws_size,
                              hipStream_t stream) {
  const float* m[4] = { (const float*)d_in[0], (const float*)d_in[1],
                        (const float*)d_in[2], (const float*)d_in[3] };
  const float* Wqkv  = (const float*)d_in[4];
  const float* Wproj = (const float*)d_in[5];
  const float* bproj = (const float*)d_in[6];

  const size_t XIN_E  = (size_t)4 * BB * NN * CC;        // 4,194,304
  const size_t WQKV_E = (size_t)4 * 3 * CC * CC;         // 3,145,728
  const size_t WPRJ_E = (size_t)4 * CC * CC;             // 1,048,576
  const size_t QKV_E  = (size_t)4 * BB * HH * NN * DD;   // 4,194,304

  _Float16* X16   = (_Float16*)d_ws;
  _Float16* Wq16  = X16  + XIN_E;
  _Float16* Wp16  = Wq16 + WQKV_E;
  _Float16* Q     = Wp16 + WPRJ_E;
  _Float16* K     = Q    + QKV_E;
  _Float16* Vt    = K    + QKV_E;
  _Float16* Xattn = Vt   + QKV_E;

  const size_t MOD_E = (size_t)BB * NN * CC;
  for (int i = 0; i < 4; ++i)
    cvt_kernel<<<(int)(MOD_E / 4 / 256), 256, 0, stream>>>(m[i], X16 + i * MOD_E,
                                                           (int)(MOD_E / 4));
  cvt_kernel<<<(int)(WQKV_E / 4 / 256), 256, 0, stream>>>(Wqkv, Wq16, (int)(WQKV_E / 4));
  cvt_kernel<<<(int)(WPRJ_E / 4 / 256), 256, 0, stream>>>(Wproj, Wp16, (int)(WPRJ_E / 4));

  qkv_kernel<<<1536, 256, 0, stream>>>(X16, Wq16, Q, K, Vt);   // 12288 waves
  attn_kernel<<<256, 256, 0, stream>>>(Q, K, Vt, Xattn);       // 2048 waves
  proj_kernel<<<512, 256, 0, stream>>>(Xattn, Wp16, bproj, (float*)d_out); // 4096 waves
}
